// SpanExtractionPooler_32152125178211
// MI455X (gfx1250) — matrix operand order repaired and checked by reading, not compiled, not run on hardware
//
#include <hip/hip_runtime.h>
#include <hip/hip_bf16.h>
#include <math.h>

typedef __attribute__((ext_vector_type(16))) __bf16 v16bf;
typedef __attribute__((ext_vector_type(8)))  __bf16 v8bf;
typedef __attribute__((ext_vector_type(8)))  float  v8f;
typedef __attribute__((ext_vector_type(4)))  int    v4i;

typedef __attribute__((address_space(1))) v4i* as1_v4i_ptr;
typedef __attribute__((address_space(3))) v4i* as3_v4i_ptr;

#define B_    8
#define S_    2048
#define H_    768
#define E_    32
#define LMAX  4
#define NSPAN (S_*LMAX)   /* 8192 spans per batch */
#define M_    (B_*S_)     /* 16384 GEMM rows      */
#define K_    H_
#define N_    H_
#define KP    128         /* K panel staged in LDS */

#if __has_builtin(__builtin_amdgcn_global_load_async_to_lds_b128)
#define HAVE_ASYNC_LDS 1
#else
#define HAVE_ASYNC_LDS 0
#endif

// ---------------------------------------------------------------------------
// K0: convert hidden_states (f32 -> bf16) and the start/end weight blocks of
//     W1 (rows 768..2303, contiguous) to bf16. Grid-stride.
// ---------------------------------------------------------------------------
__global__ void cvt_kernel(const float* __restrict__ X, const float* __restrict__ W1,
                           __bf16* __restrict__ Xb, __bf16* __restrict__ Wse)
{
    const long long nx = (long long)M_ * K_;
    const long long nw = 2LL * H_ * H_;
    long long i      = (long long)blockIdx.x * blockDim.x + threadIdx.x;
    long long stride = (long long)gridDim.x * blockDim.x;
    for (; i < nx + nw; i += stride) {
        if (i < nx) Xb[i] = (__bf16)X[i];
        else        Wse[i - nx] = (__bf16)W1[(long long)H_ * H_ + (i - nx)];
    }
}

// ---------------------------------------------------------------------------
// K1: tiny precomputes in f32:
//     Ptop[b][j] = b1[j] + sum_k topic[b][k] * W1[k][j]        (rows 0..767)
//     Plen[l][j] =         sum_k lemb[l][k]  * W1[2304+k][j]   (rows 2304..2335)
// ---------------------------------------------------------------------------
__global__ void pre_kernel(const float* __restrict__ topic, const float* __restrict__ lemb,
                           const float* __restrict__ W1, const float* __restrict__ b1,
                           float* __restrict__ Ptop, float* __restrict__ Plen)
{
    const int j   = blockIdx.x * blockDim.x + threadIdx.x;
    const int row = blockIdx.y;
    if (j >= H_) return;
    if (row < B_) {
        float acc = b1[j];
        const float* t = topic + row * H_;
        for (int k = 0; k < H_; ++k) acc = fmaf(t[k], W1[k * H_ + j], acc);
        Ptop[row * H_ + j] = acc;
    } else {
        const int l = row - B_;
        float acc = 0.0f;
        const float* e = lemb + l * E_;
        const float* w = W1 + (size_t)(3 * H_) * H_;
        for (int k = 0; k < E_; ++k) acc = fmaf(e[k], w[k * H_ + j], acc);
        Plen[l * H_ + j] = acc;
    }
}

// ---------------------------------------------------------------------------
// K2: fused dual GEMM, WMMA bf16 -> f32, async-LDS staged A panels.
//     Ps = Xb @ Ws, Pe = Xb @ We   (M=16384, N=768, K=768)
//     Block = 256 threads = 8 waves. Block tile: 32 M x 128 N.
//     A panel (32 x 128 bf16, 8 KB) double-buffered in LDS via
//     GLOBAL_LOAD_ASYNC_TO_LDS_B128 (ASYNCcnt), prefetching panel p+1
//     while WMMAs consume panel p. Each wave: 2 M-subtiles x 16 N x 2
//     matrices = 4 v8f accumulators; each B fragment feeds 2 WMMAs.
// ---------------------------------------------------------------------------
__global__ void __launch_bounds__(256)
gemm_dual_kernel(const __bf16* __restrict__ Xb, const __bf16* __restrict__ Ws,
                 const __bf16* __restrict__ We, float* __restrict__ Ps,
                 float* __restrict__ Pe)
{
    __shared__ __bf16 smem[2][32][KP];          // 2 x 8 KB A panels

    const int tid   = threadIdx.x;
    const int lane  = tid & 31;
    const int wave  = tid >> 5;
    const int nblk  = blockIdx.x * 128;         // 6 N blocks
    const int mbase = blockIdx.y * 32;          // 512 M blocks
    const int n0    = nblk + wave * 16;
    const int mrow  = lane & 15;
    const int h     = lane >> 4;                // K-half select (A) / K-group (B)

    // Issue one 32x128 A panel into LDS buffer `buf` (2 x b128 per thread).
    auto issue_panel = [&](int p, int buf) {
        const __bf16* g = Xb + (size_t)mbase * K_ + p * KP;
#pragma unroll
        for (int c = tid; c < 512; c += 256) {
            const int row = c >> 4;
            const int ch  = c & 15;
            const __bf16* gp = g + (size_t)row * K_ + ch * 8;
            __bf16* lp = &smem[buf][row][ch * 8];
#if HAVE_ASYNC_LDS
            __builtin_amdgcn_global_load_async_to_lds_b128(
                (as1_v4i_ptr)(v4i*)(__bf16*)gp,
                (as3_v4i_ptr)(v4i*)lp, 0, 0);
#else
            *(v8bf*)lp = *(const v8bf*)gp;
#endif
        }
    };
    auto commit_panel = [&]() {
#if HAVE_ASYNC_LDS
#if __has_builtin(__builtin_amdgcn_s_wait_asynccnt)
        __builtin_amdgcn_s_wait_asynccnt(0);
#else
        asm volatile("s_wait_asynccnt 0x0" ::: "memory");
#endif
#endif
        __syncthreads();
    };

    v8f cs0 = {}, ce0 = {}, cs1 = {}, ce1 = {};

    issue_panel(0, 0);
    commit_panel();

    for (int p = 0; p < K_ / KP; ++p) {
        const int cur = p & 1;
        if (p + 1 < K_ / KP) issue_panel(p + 1, cur ^ 1);

#pragma unroll
        for (int ks = 0; ks < KP / 32; ++ks) {
            const int kl = ks * 32;
            // A fragments from LDS: lane m holds K = {kl+8h..+7} and {kl+16+8h..+7}
            v8bf lo0 = *(const v8bf*)&smem[cur][mrow][kl + 8 * h];
            v8bf hi0 = *(const v8bf*)&smem[cur][mrow][kl + 16 + 8 * h];
            v8bf lo1 = *(const v8bf*)&smem[cur][16 + mrow][kl + 8 * h];
            v8bf hi1 = *(const v8bf*)&smem[cur][16 + mrow][kl + 16 + 8 * h];
            v16bf a0, a1;
#pragma unroll
            for (int i = 0; i < 8; ++i) {
                a0[i] = lo0[i]; a0[i + 8] = hi0[i];
                a1[i] = lo1[i]; a1[i + 8] = hi1[i];
            }
            // B fragments from global (W resident in WGP$/L2):
            // lane holds N-col n0+(lane&15), K = {kl+8h..+7} and {kl+16+8h..+7}
            const int kk = p * KP + kl + 8 * h;
            const int nn = n0 + (lane & 15);
            v16bf bs, be;
#pragma unroll
            for (int i = 0; i < 8; ++i) {
                bs[i]     = Ws[(size_t)(kk + i) * N_ + nn];
                bs[i + 8] = Ws[(size_t)(kk + 16 + i) * N_ + nn];
                be[i]     = We[(size_t)(kk + i) * N_ + nn];
                be[i + 8] = We[(size_t)(kk + 16 + i) * N_ + nn];
            }

            cs0 = __builtin_amdgcn_wmma_f32_16x16x32_bf16(false, a0, false, bs,
                                                          (short)0, cs0, false, false);
            cs1 = __builtin_amdgcn_wmma_f32_16x16x32_bf16(false, a1, false, bs,
                                                          (short)0, cs1, false, false);
            ce0 = __builtin_amdgcn_wmma_f32_16x16x32_bf16(false, a0, false, be,
                                                          (short)0, ce0, false, false);
            ce1 = __builtin_amdgcn_wmma_f32_16x16x32_bf16(false, a1, false, be,
                                                          (short)0, ce1, false, false);
        }
        if (p + 1 < K_ / KP) commit_panel();
    }

    // D layout: VGPR r -> M = r (+8 for lanes 16..31), lane&15 -> N
    const int n   = n0 + (lane & 15);
    const int mb0 = mbase + 8 * h;
    const int mb1 = mbase + 16 + 8 * h;
#pragma unroll
    for (int r = 0; r < 8; ++r) {
        Ps[(size_t)(mb0 + r) * N_ + n] = cs0[r];
        Pe[(size_t)(mb0 + r) * N_ + n] = ce0[r];
        Ps[(size_t)(mb1 + r) * N_ + n] = cs1[r];
        Pe[(size_t)(mb1 + r) * N_ + n] = ce1[r];
    }
}

// ---------------------------------------------------------------------------
// K3: span epilogue. One block per (b, s); wave w handles span offset w.
//     base = Ptop[b] + Ps[b,s] computed once in LDS (shared by all 4 waves);
//     score = sum_j gelu(base + Pe + Plen) * W2[j] + b2; invalid -> -inf.
// ---------------------------------------------------------------------------
__global__ void __launch_bounds__(128)
span_kernel(const float* __restrict__ Ptop, const float* __restrict__ Ps,
            const float* __restrict__ Pe, const float* __restrict__ Plen,
            const float* __restrict__ W2, const float* __restrict__ b2,
            float* __restrict__ span_out)
{
    __shared__ float base[H_];
    const int s    = blockIdx.x;
    const int b    = blockIdx.y;
    const int off  = threadIdx.x >> 5;   // wave id = span offset 0..3
    const int lane = threadIdx.x & 31;
    const int e    = min(s + off, S_ - 1);

    const float* pt = Ptop + b * H_;
    const float* ps = Ps + ((size_t)b * S_ + s) * H_;
    for (int j = threadIdx.x; j < H_; j += 128) base[j] = pt[j] + ps[j];
    __syncthreads();

    const float* pe = Pe + ((size_t)b * S_ + e) * H_;
    const float* pl = Plen + (off + 1) * H_;

    float acc = 0.0f;
    for (int j = lane; j < H_; j += 32) {
        float x = base[j] + pe[j] + pl[j];
        float g = 0.5f * x * (1.0f + erff(x * 0.70710678118654752f)); // exact gelu
        acc = fmaf(g, W2[j], acc);
    }
#pragma unroll
    for (int o = 16; o > 0; o >>= 1) acc += __shfl_xor(acc, o, 32);

    if (lane == 0) {
        float sc = acc + b2[0];
        span_out[(size_t)b * NSPAN + s * LMAX + off] =
            (s + off < S_) ? sc : -INFINITY;
    }
}

// ---------------------------------------------------------------------------
// K4: token max over the <=10 covering spans, then apply text mask.
// ---------------------------------------------------------------------------
__global__ void token_kernel(const float* __restrict__ span,
                             const unsigned char* __restrict__ mask,
                             float* __restrict__ tok)
{
    const int i = blockIdx.x * blockDim.x + threadIdx.x;
    if (i >= B_ * S_) return;
    const int b = i / S_;
    const int t = i % S_;
    const float* sp = span + (size_t)b * NSPAN;
    float m = -INFINITY;
#pragma unroll
    for (int off = 0; off < LMAX; ++off) {
        int s_lo = t - off;
        if (s_lo < 0) s_lo = 0;
        for (int s = s_lo; s <= t; ++s)
            m = fmaxf(m, sp[s * LMAX + off]);
    }
    tok[i] = mask[i] ? m : -INFINITY;
}

// ---------------------------------------------------------------------------
extern "C" void kernel_launch(void* const* d_in, const int* in_sizes, int n_in,
                              void* d_out, int out_size, void* d_ws, size_t ws_size,
                              hipStream_t stream)
{
    const float*         X     = (const float*)d_in[0];
    const float*         topic = (const float*)d_in[1];
    const unsigned char* mask  = (const unsigned char*)d_in[2];
    const float*         lemb  = (const float*)d_in[3];
    const float*         W1    = (const float*)d_in[4];
    const float*         b1    = (const float*)d_in[5];
    const float*         W2    = (const float*)d_in[6];
    const float*         b2    = (const float*)d_in[7];

    float* tok_out  = (float*)d_out;            // (B, S) first
    float* span_out = tok_out + B_ * S_;        // then (B, N)

    // Workspace layout (~128 MB total)
    char* ws = (char*)d_ws;
    __bf16* Xb  = (__bf16*)ws;  ws += (size_t)M_ * K_ * 2;       // 25.2 MB
    __bf16* Wse = (__bf16*)ws;  ws += (size_t)2 * H_ * H_ * 2;   //  2.4 MB
    float*  Ps  = (float*)ws;   ws += (size_t)M_ * N_ * 4;       // 50.3 MB
    float*  Pe  = (float*)ws;   ws += (size_t)M_ * N_ * 4;       // 50.3 MB
    float*  Ptop= (float*)ws;   ws += (size_t)B_ * H_ * 4;
    float*  Plen= (float*)ws;

    cvt_kernel <<<2048, 256, 0, stream>>>(X, W1, Xb, Wse);
    pre_kernel <<<dim3((H_ + 255) / 256, B_ + LMAX + 1), 256, 0, stream>>>(
        topic, lemb, W1, b1, Ptop, Plen);
    gemm_dual_kernel<<<dim3(N_ / 128, M_ / 32), 256, 0, stream>>>(
        Xb, Wse, Wse + (size_t)H_ * H_, Ps, Pe);
    span_kernel<<<dim3(S_, B_), 128, 0, stream>>>(Ptop, Ps, Pe, Plen, W2, b2, span_out);
    token_kernel<<<(B_ * S_ + 255) / 256, 256, 0, stream>>>(span_out, mask, tok_out);
}
